// GAT_49589692400146
// MI455X (gfx1250) — compile-verified, matrix operand order checked by reading, not run on hardware
//
#include <hip/hip_runtime.h>
#include <hip/hip_bf16.h>

#define NN 6144
#define NF 512
#define NH 256
#define ALPHA 0.2f
#define NEGINF -9.0e15f
#define CHUNK 64

typedef __attribute__((ext_vector_type(16))) __bf16 v16bf;
typedef __attribute__((ext_vector_type(8)))  float  v8f;
typedef __attribute__((ext_vector_type(4)))  int    i4vec;   // clang ext vector
typedef __attribute__((ext_vector_type(4)))  float  f4vec;   // (builtin-friendly)

union AFrag { v16bf v; uint4 u[2]; };
union CFrag { v8f  v; float f[8]; };

__device__ __forceinline__ unsigned short f2bf(float f) {
  unsigned int b = __float_as_uint(f);
  b += 0x7FFFu + ((b >> 16) & 1u);          // round-to-nearest-even
  return (unsigned short)(b >> 16);
}
__device__ __forceinline__ float bf2f(unsigned short h) {
  return __uint_as_float(((unsigned int)h) << 16);
}
__device__ __forceinline__ uint4 pack8(const float* f) {
  uint4 r;
  r.x = f2bf(f[0]) | ((unsigned)f2bf(f[1]) << 16);
  r.y = f2bf(f[2]) | ((unsigned)f2bf(f[3]) << 16);
  r.z = f2bf(f[4]) | ((unsigned)f2bf(f[5]) << 16);
  r.w = f2bf(f[6]) | ((unsigned)f2bf(f[7]) << 16);
  return r;
}
__device__ __forceinline__ float lrelu(float e) { return e > 0.f ? e : ALPHA * e; }

// ---------------------------------------------------------------- prep:
// xb  = bf16(x)        row-major [6144][512]  (A-operand friendly)
// Wt  = bf16(W^T)      [256][512]             (B-operand friendly)
__global__ void __launch_bounds__(256) gat_prep(
    const float* __restrict__ x, const float* __restrict__ W,
    unsigned short* __restrict__ xb, unsigned short* __restrict__ Wt) {
  const int bid = blockIdx.x, tid = threadIdx.x;
  if (bid < 12288) {                       // 12288*256 = 6144*512
    size_t idx = (size_t)bid * 256 + tid;
    xb[idx] = f2bf(__builtin_nontemporal_load(x + idx));
  } else {                                 // 512*256 W elements
    int idx = (bid - 12288) * 256 + tid;
    int k = idx >> 8, c = idx & 255;       // W row-major [512][256]
    Wt[(size_t)c * NF + k] = f2bf(W[idx]);
  }
}

// ---------------------------------------------------------------- gemm1:
// WhT = bf16( (x@W)^T )  [256][6144]   via v_wmma_f32_16x16x32_bf16
__global__ void __launch_bounds__(256) gat_gemm1(
    const unsigned short* __restrict__ xb, const unsigned short* __restrict__ Wt,
    unsigned short* __restrict__ WhT) {
  const int r0   = blockIdx.x * 16;
  const int lane = threadIdx.x & 31;
  const int wave = threadIdx.x >> 5;
  const int col0 = wave * 32;
  const int lr   = lane & 15;
  const int koff = (lane >> 4) << 3;       // 0 or 8 (fragment half-select)

  CFrag acc0, acc1;
#pragma unroll
  for (int v = 0; v < 8; ++v) { acc0.f[v] = 0.f; acc1.f[v] = 0.f; }

  const unsigned short* xrow = xb + (size_t)(r0 + lr) * NF + koff;
  const unsigned short* wt0  = Wt + (size_t)(col0 + lr) * NF + koff;
  const unsigned short* wt1  = Wt + (size_t)(col0 + 16 + lr) * NF + koff;

  for (int k0 = 0; k0 < NF; k0 += 32) {
    AFrag A, B0, B1;
    A.u[0]  = *(const uint4*)(xrow + k0);
    A.u[1]  = *(const uint4*)(xrow + k0 + 16);
    B0.u[0] = *(const uint4*)(wt0 + k0);
    B0.u[1] = *(const uint4*)(wt0 + k0 + 16);
    B1.u[0] = *(const uint4*)(wt1 + k0);
    B1.u[1] = *(const uint4*)(wt1 + k0 + 16);
    acc0.v = __builtin_amdgcn_wmma_f32_16x16x32_bf16(false, A.v, false, B0.v,
                                                     (short)0, acc0.v, false, false);
    acc1.v = __builtin_amdgcn_wmma_f32_16x16x32_bf16(false, A.v, false, B1.v,
                                                     (short)0, acc1.v, false, false);
  }
  // transposed bf16 store: lane = fixed column, VGPR v = row -> contiguous 16B
  *(uint4*)&WhT[(size_t)(col0 + lr)      * NN + r0 + koff] = pack8(acc0.f);
  *(uint4*)&WhT[(size_t)(col0 + 16 + lr) * NN + r0 + koff] = pack8(acc1.f);
}

// ---------------------------------------------------------------- e1/e2:
__global__ void __launch_bounds__(256) gat_e12(
    const unsigned short* __restrict__ WhT, const float* __restrict__ a,
    float* __restrict__ e1, float* __restrict__ e2) {
  const int i = blockIdx.x * 256 + threadIdx.x;
  float s1 = 0.f, s2 = 0.f;
  for (int c = 0; c < NH; ++c) {
    float w = bf2f(WhT[(size_t)c * NN + i]);   // coalesced across threads
    s1 += w * a[c];
    s2 += w * a[NH + c];
  }
  e1[i] = s1; e2[i] = s2;
}

// ---------------------------------------------------------------- flash:
// one pass over adj (b128 nontemporal); online softmax; p(bf16) @ Wh via
// WMMA; fused elu / fcW row-dot / deterministic column sums.
__global__ void __launch_bounds__(256) gat_flash(
    const int* __restrict__ adj,
    const float* __restrict__ e1g, const float* __restrict__ e2g,
    const unsigned short* __restrict__ WhT, const float* __restrict__ fcW,
    float* __restrict__ out, float* __restrict__ s_part) {
  __shared__ __align__(16) unsigned short pt[16 * CHUNK];  // p tile, A layout
  __shared__ float sh_m[16], sh_l[16], sh_scale[16], sh_cm[16];
  __shared__ float outw[8 * 16];

  const int tid  = threadIdx.x;
  const int r0   = blockIdx.x * 16;
  const int r    = tid >> 4;                 // 0..15 row within tile
  const int q    = tid & 15;                 // owns j-offsets 4q..4q+3
  const int lane = tid & 31;
  const int wave = tid >> 5;
  const int col0 = wave * 32;
  const int lr   = lane & 15;
  const int koff = (lane >> 4) << 3;

  if (tid < 16) { sh_m[tid] = -3.0e38f; sh_l[tid] = 0.f; }

  CFrag acc0, acc1;
#pragma unroll
  for (int v = 0; v < 8; ++v) { acc0.f[v] = 0.f; acc1.f[v] = 0.f; }

  const float e1v = e1g[r0 + r];
  const int* adjrow = adj + (size_t)(r0 + r) * NN + 4 * q;
  const float* e2p  = e2g + 4 * q;
  const unsigned short* wtp0 = WhT + (size_t)(col0 + lr) * NN + koff;
  const unsigned short* wtp1 = WhT + (size_t)(col0 + 16 + lr) * NN + koff;
  __syncthreads();

  for (int j0 = 0; j0 < NN; j0 += CHUNK) {
    // ---- S1: stream adj (b128, nontemporal) + e2, compute masked leaky e
    const i4vec av = __builtin_nontemporal_load((const i4vec*)(adjrow + j0));
    const f4vec ez = *(const f4vec*)(e2p + j0);
    float ev[4];
    ev[0] = av.x > 0 ? lrelu(e1v + ez.x) : NEGINF;
    ev[1] = av.y > 0 ? lrelu(e1v + ez.y) : NEGINF;
    ev[2] = av.z > 0 ? lrelu(e1v + ez.z) : NEGINF;
    ev[3] = av.w > 0 ? lrelu(e1v + ez.w) : NEGINF;

    float cmax = fmaxf(fmaxf(ev[0], ev[1]), fmaxf(ev[2], ev[3]));
    cmax = fmaxf(cmax, __shfl_xor(cmax, 1, 32));
    cmax = fmaxf(cmax, __shfl_xor(cmax, 2, 32));
    cmax = fmaxf(cmax, __shfl_xor(cmax, 4, 32));
    cmax = fmaxf(cmax, __shfl_xor(cmax, 8, 32));
    if (q == 0) sh_cm[r] = cmax;
    __syncthreads();

    // ---- S2: online-softmax bookkeeping
    if (tid < 16) {
      float nm = fmaxf(sh_m[tid], sh_cm[tid]);
      float sc = __expf(sh_m[tid] - nm);
      sh_scale[tid] = sc;
      sh_l[tid] *= sc;
      sh_m[tid] = nm;
    }
    __syncthreads();

    // ---- S3: p = exp(e - m) -> bf16 tile + row-sum into l
    const float m = sh_m[r];
    float p0 = __expf(ev[0] - m);
    float p1 = __expf(ev[1] - m);
    float p2 = __expf(ev[2] - m);
    float p3 = __expf(ev[3] - m);
    float ps = (p0 + p1) + (p2 + p3);
    ps += __shfl_xor(ps, 1, 32);
    ps += __shfl_xor(ps, 2, 32);
    ps += __shfl_xor(ps, 4, 32);
    ps += __shfl_xor(ps, 8, 32);
    if (q == 0) sh_l[r] += ps;
    uint2 pk;
    pk.x = f2bf(p0) | ((unsigned)f2bf(p1) << 16);
    pk.y = f2bf(p2) | ((unsigned)f2bf(p3) << 16);
    *(uint2*)&pt[r * CHUNK + 4 * q] = pk;
    __syncthreads();

    // ---- WMMA region: rescale acc, then 2 K-steps x 2 column tiles
#pragma unroll
    for (int v = 0; v < 8; ++v) {
      const float sc = sh_scale[v + koff];
      acc0.f[v] *= sc;
      acc1.f[v] *= sc;
    }
#pragma unroll
    for (int kk = 0; kk < 2; ++kk) {
      const int kb = kk * 32;
      AFrag A, B0, B1;
      A.u[0]  = *(const uint4*)&pt[lr * CHUNK + kb + koff];
      A.u[1]  = *(const uint4*)&pt[lr * CHUNK + kb + 16 + koff];
      B0.u[0] = *(const uint4*)(wtp0 + j0 + kb);
      B0.u[1] = *(const uint4*)(wtp0 + j0 + kb + 16);
      B1.u[0] = *(const uint4*)(wtp1 + j0 + kb);
      B1.u[1] = *(const uint4*)(wtp1 + j0 + kb + 16);
      acc0.v = __builtin_amdgcn_wmma_f32_16x16x32_bf16(false, A.v, false, B0.v,
                                                       (short)0, acc0.v, false, false);
      acc1.v = __builtin_amdgcn_wmma_f32_16x16x32_bf16(false, A.v, false, B1.v,
                                                       (short)0, acc1.v, false, false);
    }
    // no trailing barrier needed: next iteration's S1/S2 barriers order the
    // pt/sh_scale reads here against their next writes (which occur post-S1).
  }
  __syncthreads();                           // close last WMMA region

  // --------------------------- epilogue: h = elu(acc/l); out-dot; col sums
  if (tid < 16) sh_scale[tid] = 1.0f / sh_l[tid];
  __syncthreads();

  const float w0 = fcW[col0 + lr];
  const float w1 = fcW[col0 + 16 + lr];
  float po[8];
  float ss0 = 0.f, ss1 = 0.f;
#pragma unroll
  for (int v = 0; v < 8; ++v) {
    const float inv = sh_scale[v + koff];
    float h0 = acc0.f[v] * inv;
    h0 = h0 > 0.f ? h0 : (__expf(h0) - 1.f);
    float h1 = acc1.f[v] * inv;
    h1 = h1 > 0.f ? h1 : (__expf(h1) - 1.f);
    ss0 += h0;
    ss1 += h1;
    po[v] = h0 * w0 + h1 * w1;
  }
#pragma unroll
  for (int v = 0; v < 8; ++v) {              // reduce over 16 columns/lane-group
    po[v] += __shfl_xor(po[v], 1, 32);
    po[v] += __shfl_xor(po[v], 2, 32);
    po[v] += __shfl_xor(po[v], 4, 32);
    po[v] += __shfl_xor(po[v], 8, 32);
  }
  if (lr == 0) {
#pragma unroll
    for (int v = 0; v < 8; ++v) outw[wave * 16 + koff + v] = po[v];
  }
  // deterministic per-block column sums of elu(h)
  ss0 += __shfl_xor(ss0, 16, 32);
  ss1 += __shfl_xor(ss1, 16, 32);
  if (lane < 16) {
    s_part[(size_t)blockIdx.x * NH + col0 + lr]      = ss0;
    s_part[(size_t)blockIdx.x * NH + col0 + 16 + lr] = ss1;
  }
  __syncthreads();
  if (tid < 16) {
    float o = 0.f;
#pragma unroll
    for (int w = 0; w < 8; ++w) o += outw[w * 16 + tid];
    out[r0 + tid] = o;                       // partial: missing scalar term
  }
}

// ---------------------------------------------------------------- s reduce:
__global__ void __launch_bounds__(256) gat_sreduce(
    const float* __restrict__ s_part, float* __restrict__ s) {
  const int c = threadIdx.x;
  float acc = 0.f;
  for (int b = 0; b < 384; ++b) acc += s_part[(size_t)b * NH + c];
  s[c] = acc;
}

// ---------------------------------------------------------------- final:
// out[i] += dot(s, fcW[256:512]) + fcb   (scalar, same for every row)
__global__ void __launch_bounds__(256) gat_final(
    const float* __restrict__ s, const float* __restrict__ fcW,
    const float* __restrict__ fcb, float* __restrict__ out) {
  float dot = fcb[0];
  for (int c = 0; c < NH; ++c) dot += s[c] * fcW[NH + c];
  const int i = blockIdx.x * 256 + threadIdx.x;
  out[i] += dot;
}

extern "C" void kernel_launch(void* const* d_in, const int* in_sizes, int n_in,
                              void* d_out, int out_size, void* d_ws, size_t ws_size,
                              hipStream_t stream) {
  const float* x   = (const float*)d_in[0];
  const int*   adj = (const int*)d_in[1];
  const float* W   = (const float*)d_in[2];
  const float* a   = (const float*)d_in[3];
  const float* fcW = (const float*)d_in[4];
  const float* fcb = (const float*)d_in[5];
  float* out = (float*)d_out;

  char* ws = (char*)d_ws;
  unsigned short* xb     = (unsigned short*)(ws);                 // 6,291,456 B
  unsigned short* Wt     = (unsigned short*)(ws + 6291456);       //   262,144 B
  unsigned short* WhT    = (unsigned short*)(ws + 6553600);       // 3,145,728 B
  float*          e1     = (float*)(ws + 9699328);                //    24,576 B
  float*          e2     = (float*)(ws + 9723904);                //    24,576 B
  float*          s      = (float*)(ws + 9748480);                //     1,024 B
  float*          s_part = (float*)(ws + 9749504);                //   393,216 B

  gat_prep   <<<12800, 256, 0, stream>>>(x, W, xb, Wt);
  gat_gemm1  <<<  384, 256, 0, stream>>>(xb, Wt, WhT);
  gat_e12    <<<   24, 256, 0, stream>>>(WhT, a, e1, e2);
  gat_flash  <<<  384, 256, 0, stream>>>(adj, e1, e2, WhT, fcW, out, s_part);
  gat_sreduce<<<    1, 256, 0, stream>>>(s_part, s);
  gat_final  <<<   24, 256, 0, stream>>>(s, fcW, fcb, out);
}
